// GCNLayer_lstm_22565758173854
// MI455X (gfx1250) — matrix operand order repaired and checked by reading, not compiled, not run on hardware
//
#include <hip/hip_runtime.h>

// ---------------- problem constants ----------------
#define NNODES   200000
#define NTILES   12500      // NNODES / 16
#define DMAX     32
#define FDIM     128
#define HDIM     128
#define KTOT     256        // F + H (fused A = [x | h])
#define ROWSTRIDE 264       // 256 cols + 8 halves pad (bank spread), ushort units
#define OUTSTRIDE 132       // 128 + 4 pad, float units
#define WS_GATE_HALFS 131072 // 32 ct * 8 kc * 32 lanes * 16 halves
#define WS_LIN_HALFS  16384  //  8 ct * 4 kc * 32 lanes * 16 halves

typedef __bf16 bf16_t;
typedef bf16_t v16bf __attribute__((ext_vector_type(16)));
typedef float  v8f   __attribute__((ext_vector_type(8)));
typedef float  f32x4 __attribute__((ext_vector_type(4)));
typedef unsigned int u32x4 __attribute__((ext_vector_type(4)));

union Frag32 { u32x4 u[2]; v16bf v; };

__device__ __forceinline__ unsigned short f2bf(float f) {
    unsigned u = __builtin_bit_cast(unsigned, f);
    return (unsigned short)((u + 0x7FFFu + ((u >> 16) & 1u)) >> 16);
}
__device__ __forceinline__ float bf2f(unsigned short b) {
    return __builtin_bit_cast(float, ((unsigned)b) << 16);
}
// hardware v_rcp_f32 (1 ULP) instead of IEEE div chain
__device__ __forceinline__ float frcp_(float x) {
    return __builtin_amdgcn_rcpf(x);
}
// single-instruction packed f32->bf16 when available (v_cvt_pk_bf16_f32)
__device__ __forceinline__ unsigned pk_bf16(float lo, float hi) {
#if __has_builtin(__builtin_amdgcn_cvt_pk_bf16_f32)
    auto p = __builtin_amdgcn_cvt_pk_bf16_f32(lo, hi);
    unsigned u;
    __builtin_memcpy(&u, &p, sizeof(u));
    return u;
#else
    return (unsigned)f2bf(lo) | ((unsigned)f2bf(hi) << 16);
#endif
}
__device__ __forceinline__ float sigmoid_(float x) {
    return 1.0f / (1.0f + __expf(-x));
}
__device__ __forceinline__ float tanh_(float x) {
    // 1 - 2/(e^{2x}+1): saturates correctly at +/-1 for large |x|
    return 1.0f - 2.0f / (__expf(2.0f * x) + 1.0f);
}
__device__ __forceinline__ unsigned scale_pk2(unsigned p, float s) {
    float lo = bf2f((unsigned short)(p & 0xFFFFu)) * s;
    float hi = bf2f((unsigned short)(p >> 16)) * s;
    return pk_bf16(lo, hi);
}

// ---------------------------------------------------------------------------
// Prep: repack W_ih/W_hh (fused K-wise) and W_lin into bf16 WMMA B-fragments.
// B 32x16 bf16, contiguous k order used here (wave32): lane<16 -> N=lane,  K = e      (e=0..15)
//                               lane>=16 -> N=lane-16, K = 16+e
// ws[( (ct*KC + kc)*32 + lane )*16 + e]
// ---------------------------------------------------------------------------
__global__ void lstm_gcn_prep(const float* __restrict__ W_ih,
                              const float* __restrict__ W_hh,
                              const float* __restrict__ W_lin,
                              unsigned short* __restrict__ ws) {
    int idx = blockIdx.x * blockDim.x + threadIdx.x;
    if (idx < WS_GATE_HALFS) {
        int e = idx & 15, lane = (idx >> 4) & 31, kc = (idx >> 9) & 7, ct = idx >> 12;
        int K = ((lane >> 4) << 4) + e;
        int Ncol = lane & 15;
        int kg = kc * 32 + K;          // fused K in [0,256)
        int ng = ct * 16 + Ncol;       // gate column in [0,512)
        float v = (kg < FDIM) ? W_ih[ng * FDIM + kg]
                              : W_hh[ng * HDIM + (kg - FDIM)];
        ws[idx] = f2bf(v);
    } else if (idx < WS_GATE_HALFS + WS_LIN_HALFS) {
        int j = idx - WS_GATE_HALFS;
        int e = j & 15, lane = (j >> 4) & 31, kc = (j >> 9) & 3, ct = j >> 11;
        int K = ((lane >> 4) << 4) + e;
        int Ncol = lane & 15;
        int kg = kc * 32 + K;          // K in [0,128)
        int ng = ct * 16 + Ncol;       // out column in [0,128)
        ws[WS_GATE_HALFS + j] = f2bf(W_lin[ng * FDIM + kg]);
    }
}

// ---------------------------------------------------------------------------
// Main: 8 waves / block, 16 nodes / tile, grid-stride over 12500 tiles.
// Wave w owns gate column tiles {w, w+8, w+16, w+24}  (= i,f,g,o slices of
// hidden columns [16w,16w+16)), so the LSTM cell update is wave-local.
// ---------------------------------------------------------------------------
__global__ __launch_bounds__(256)
void lstm_gcn_main(const float* __restrict__ feat,
                   const float* __restrict__ in_norm,
                   const float* __restrict__ out_norm,
                   const float* __restrict__ b_ih,
                   const float* __restrict__ b_hh,
                   const float* __restrict__ b_lin,
                   const int*   __restrict__ neighbors,
                   const int*   __restrict__ degrees,
                   const unsigned short* __restrict__ ws,
                   float* __restrict__ out) {
    __shared__ unsigned short ldsA[16 * ROWSTRIDE]; // 16 rows x [x(128)|h(128)] bf16, padded
    __shared__ float ldsOut[16 * OUTSTRIDE];        // f32 output tile staging
    __shared__ int   s_deg[16];
    __shared__ float s_inv_in[16];
    __shared__ int   s_dmax;

    const int tid  = threadIdx.x;
    const int lane = tid & 31;
    const int w    = tid >> 5;      // wave id 0..7
    const int hl   = lane >> 4;     // lane half
    const int nl   = lane & 15;     // N within tile / row within half

    // --- load persistent B fragments (weights) into VGPRs ---
    v16bf Bg[4][8];
#pragma unroll
    for (int g = 0; g < 4; ++g) {
        const int ct = g * 8 + w;
#pragma unroll
        for (int kc = 0; kc < 8; ++kc) {
            const unsigned short* p = ws + (((ct * 8 + kc) * 32 + lane) << 4);
            Frag32 f;
            f.u[0] = *(const u32x4*)(p);
            f.u[1] = *(const u32x4*)(p + 8);
            Bg[g][kc] = f.v;
        }
    }
    v16bf Bl[4];
#pragma unroll
    for (int kc = 0; kc < 4; ++kc) {
        const unsigned short* p = ws + WS_GATE_HALFS + (((w * 4 + kc) * 32 + lane) << 4);
        Frag32 f;
        f.u[0] = *(const u32x4*)(p);
        f.u[1] = *(const u32x4*)(p + 8);
        Bl[kc] = f.v;
    }

    float bias[4];
#pragma unroll
    for (int g = 0; g < 4; ++g)
        bias[g] = b_ih[(g * 8 + w) * 16 + nl] + b_hh[(g * 8 + w) * 16 + nl];
    const float biasL = b_lin[w * 16 + nl];

    const int gr = tid >> 4;         // gather row 0..15
    const int gc = (tid & 15) * 8;   // gather col base 0..120

    for (int tile = blockIdx.x; tile < NTILES; tile += gridDim.x) {
        const int nodeBase = tile * 16;

        if (tid < 16) {
            s_deg[tid]    = degrees[nodeBase + tid];
            s_inv_in[tid] = frcp_(in_norm[nodeBase + tid]);
        }
        { // zero h region (cols 128..255)
            u32x4 z = {0u, 0u, 0u, 0u};
            *(u32x4*)(&ldsA[gr * ROWSTRIDE + FDIM + gc]) = z;
        }
        __syncthreads();
        if (tid == 0) {
            int m = 0;
            for (int i = 0; i < 16; ++i) m = max(m, s_deg[i]);
            s_dmax = m;
        }
        __syncthreads();
        const int dmax = s_dmax;

        int myDeg[8];
#pragma unroll
        for (int r = 0; r < 8; ++r) myDeg[r] = s_deg[r + 8 * hl];

        v8f cst;
#pragma unroll
        for (int r = 0; r < 8; ++r) cst[r] = 0.0f;

        for (int t = 0; t < dmax; ++t) {
            // ---- gather x = feat[nbr] / out_norm[nbr] -> LDS bf16 (cols 0..127)
            {
                const int nbr = neighbors[(nodeBase + gr) * DMAX + t];
                const float invo = frcp_(out_norm[nbr]);
                const float* fp = feat + (size_t)nbr * FDIM + gc;
                f32x4 a0 = *(const f32x4*)(fp);
                f32x4 a1 = *(const f32x4*)(fp + 4);
                u32x4 pk;
                pk.x = pk_bf16(a0.x * invo, a0.y * invo);
                pk.y = pk_bf16(a0.z * invo, a0.w * invo);
                pk.z = pk_bf16(a1.x * invo, a1.y * invo);
                pk.w = pk_bf16(a1.z * invo, a1.w * invo);
                *(u32x4*)(&ldsA[gr * ROWSTRIDE + gc]) = pk;
            }
            __syncthreads();

            // ---- preload all 8 A fragments (minimal dscnt waits, dense WMMA)
            // A 16x32 bf16, contiguous k order used here (same as B): lane<16 K{0..15}; lane>=16 K{16..31}
            const int arow = nl * ROWSTRIDE + hl * 16;
            Frag32 af[8];
#pragma unroll
            for (int kc = 0; kc < 8; ++kc) {
                const int off0 = arow + kc * 32;
                af[kc].u[0] = *(const u32x4*)(&ldsA[off0]);
                af[kc].u[1] = *(const u32x4*)(&ldsA[off0 + 8]);
            }

            // ---- gates = [x|h] @ [W_ih;W_hh]^T + (b_ih+b_hh): 4 indep acc chains
            v8f acc[4];
#pragma unroll
            for (int g = 0; g < 4; ++g) {
#pragma unroll
                for (int e = 0; e < 8; ++e) acc[g][e] = bias[g];
            }
#pragma unroll
            for (int kc = 0; kc < 8; ++kc) {
#pragma unroll
                for (int g = 0; g < 4; ++g)
                    acc[g] = __builtin_amdgcn_wmma_f32_16x16x32_bf16(
                        false, af[kc].v, false, Bg[g][kc], (short)0, acc[g], false, false);
            }
            __syncthreads();

            // ---- LSTM cell update (wave-local i,f,g,o), masked by degree
#pragma unroll
            for (int r = 0; r < 8; ++r) {
                float iv = sigmoid_(acc[0][r]);
                float fv = sigmoid_(acc[1][r]);
                float gv = tanh_(acc[2][r]);
                float ov = sigmoid_(acc[3][r]);
                float cn = fv * cst[r] + iv * gv;
                float hn = ov * tanh_(cn);
                if (t < myDeg[r]) {
                    cst[r] = cn;
                    ldsA[(r + 8 * hl) * ROWSTRIDE + FDIM + w * 16 + nl] = f2bf(hn);
                }
            }
        }
        __syncthreads();

        // ---- scale h by 1/in_norm into cols 0..127
        {
            u32x4 hv = *(const u32x4*)(&ldsA[gr * ROWSTRIDE + FDIM + gc]);
            const float s = s_inv_in[gr];
            u32x4 pk;
            pk.x = scale_pk2(hv.x, s);
            pk.y = scale_pk2(hv.y, s);
            pk.z = scale_pk2(hv.z, s);
            pk.w = scale_pk2(hv.w, s);
            *(u32x4*)(&ldsA[gr * ROWSTRIDE + gc]) = pk;
        }
        __syncthreads();

        // ---- out = h_scaled @ W_lin^T + b_lin (wave w -> out cols [16w,16w+16))
        v8f accL;
#pragma unroll
        for (int e = 0; e < 8; ++e) accL[e] = biasL;
        {
            const int arow = nl * ROWSTRIDE + hl * 16;
            Frag32 afl[4];
#pragma unroll
            for (int kc = 0; kc < 4; ++kc) {
                const int off0 = arow + kc * 32;
                afl[kc].u[0] = *(const u32x4*)(&ldsA[off0]);
                afl[kc].u[1] = *(const u32x4*)(&ldsA[off0 + 8]);
            }
#pragma unroll
            for (int kc = 0; kc < 4; ++kc)
                accL = __builtin_amdgcn_wmma_f32_16x16x32_bf16(
                    false, afl[kc].v, false, Bl[kc], (short)0, accL, false, false);
        }

        // ---- stage output tile in LDS, then coalesced b128 stores
#pragma unroll
        for (int r = 0; r < 8; ++r)
            ldsOut[(r + 8 * hl) * OUTSTRIDE + w * 16 + nl] = accL[r];
        __syncthreads();
        {
            f32x4 o0 = *(const f32x4*)(&ldsOut[gr * OUTSTRIDE + gc]);
            f32x4 o1 = *(const f32x4*)(&ldsOut[gr * OUTSTRIDE + gc + 4]);
            float* op = out + (size_t)(nodeBase + gr) * 128 + gc;
            *(f32x4*)(op)     = o0;
            *(f32x4*)(op + 4) = o1;
        }
        __syncthreads(); // all LDS consumers done before next tile reuses buffers
    }
}

// ---------------------------------------------------------------------------
extern "C" void kernel_launch(void* const* d_in, const int* in_sizes, int n_in,
                              void* d_out, int out_size, void* d_ws, size_t ws_size,
                              hipStream_t stream) {
    const float* feat      = (const float*)d_in[0];
    const float* in_norm   = (const float*)d_in[1];
    const float* out_norm  = (const float*)d_in[2];
    const float* W_ih      = (const float*)d_in[3];
    const float* W_hh      = (const float*)d_in[4];
    const float* b_ih      = (const float*)d_in[5];
    const float* b_hh      = (const float*)d_in[6];
    const float* W_lin     = (const float*)d_in[7];
    const float* b_lin     = (const float*)d_in[8];
    const int*   neighbors = (const int*)d_in[9];
    const int*   degrees   = (const int*)d_in[10];
    float* out = (float*)d_out;
    unsigned short* ws = (unsigned short*)d_ws;

    const int prepElems = WS_GATE_HALFS + WS_LIN_HALFS;
    lstm_gcn_prep<<<(prepElems + 255) / 256, 256, 0, stream>>>(W_ih, W_hh, W_lin, ws);
    lstm_gcn_main<<<1280, 256, 0, stream>>>(feat, in_norm, out_norm,
                                            b_ih, b_hh, b_lin,
                                            neighbors, degrees, ws, out);
}
